// SparseGlobalAttention_19756849562023
// MI455X (gfx1250) — compile-verified, matrix operand order checked by reading, not run on hardware
//
#include <hip/hip_runtime.h>
#include <hip/hip_bf16.h>
#include <stdint.h>

// ---------------------------------------------------------------------------
// Sparse global attention for MI455X (gfx1250, wave32):
//   WMMA bf16 16x16x32 for all GEMMs + attention, double-buffered TDM
//   (tensor_load_to_lds) tile staging, wave-pipelined flash attention with
//   batched cross-lane softmax reductions.
// ---------------------------------------------------------------------------

typedef __bf16 bf16_t;
typedef __attribute__((ext_vector_type(16))) __bf16 v16bf;
typedef __attribute__((ext_vector_type(8)))  float   v8f;
typedef __attribute__((ext_vector_type(2)))  __bf16 bf16x2;
typedef __attribute__((ext_vector_type(8)))  __bf16 bf16x8;
typedef __attribute__((ext_vector_type(4)))  unsigned int u32x4;
typedef __attribute__((ext_vector_type(8)))  int i32x8;
typedef __attribute__((ext_vector_type(4)))  int i32x4;

#define B_    8
#define L_    4096
#define D_    2048
#define H_    16
#define HD_   128
#define MK_   1024
#define KROWS 1056   // 1024 tokens + 1 register key, padded to 32-key tiles

__device__ __forceinline__ v8f wmma_bf16(v16bf a, v16bf b, v8f c) {
  // v_wmma_f32_16x16x32_bf16  D = A(16x32) * B(32x16) + C
  return __builtin_amdgcn_wmma_f32_16x16x32_bf16(false, a, false, b,
                                                 (short)0, c, false, false);
}

// ---------------------------------------------------------------------------
// TDM: post a 2D bf16 tile DMA (global -> LDS). tile dims == tensor dims (no
// OOB clipping). D# layout per CDNA5 ISA 8.3/8.4: count=1, data_size=2B,
// type=2 ("image"), tensor_dim0_stride in elements (48-bit).
// ---------------------------------------------------------------------------
__device__ __forceinline__ void tdm_load_tile_bf16(
    unsigned lds_off, const void* gaddr,
    unsigned tile_d0, unsigned tile_d1, unsigned long long d0_stride) {
  const unsigned long long ga = (unsigned long long)(uintptr_t)gaddr;
  u32x4 g0;
  g0[0] = 1u;                                     // count=1 (valid user desc)
  g0[1] = lds_off;                                // lds_addr (bytes)
  g0[2] = (unsigned)(ga & 0xffffffffull);         // global_addr[31:0]
  g0[3] = (unsigned)((ga >> 32) & 0x1ffffffull)   // global_addr[56:32]
        | (2u << 30);                             // type=2
  i32x8 g1;
  g1[0] = (int)(1u << 16);                        // data_size=1 -> 2 bytes
  g1[1] = (int)((tile_d0 & 0xffffu) << 16);       // tensor_dim0[15:0]
  g1[2] = (int)(((tile_d0 >> 16) & 0xffffu) |     // tensor_dim0[31:16]
                ((tile_d1 & 0xffffu) << 16));     // tensor_dim1[15:0]
  g1[3] = (int)(((tile_d1 >> 16) & 0xffffu) |     // tensor_dim1[31:16]
                ((tile_d0 & 0xffffu) << 16));     // tile_dim0
  g1[4] = (int)(tile_d1 & 0xffffu);               // tile_dim1 (tile_dim2=0)
  g1[5] = (int)(unsigned)(d0_stride & 0xffffffffull);    // dim0_stride[31:0]
  g1[6] = (int)(unsigned)((d0_stride >> 32) & 0xffffull);// dim0_stride[47:32]
  g1[7] = 0;                                      // tensor_dim1_stride hi = 0
  const i32x4 z4 = {0, 0, 0, 0};                  // groups 2/3 unused (2D)
  const i32x8 z8 = {0, 0, 0, 0, 0, 0, 0, 0};
  __builtin_amdgcn_tensor_load_to_lds(g0, g1, z4, z4, z8, 0);
}

__device__ __forceinline__ unsigned lds_offset_of(const void* p) {
  return (unsigned)(uintptr_t)p;   // generic LDS pointer: low 32 bits = offset
}

// Load a 16x32 A/B-operand fragment row. `rowptr` points at the 32 contiguous
// K values of this lane's row (m or n = lane&15); `hi` = lane>>4 selects the
// K half per ISA 7.12.2 (lanes 0-15: K 0-7,16-23; lanes 16-31: K 8-15,24-31).
__device__ __forceinline__ v16bf load_frag_row32(const bf16_t* rowptr, int hi) {
  v16bf f;
#pragma unroll
  for (int g = 0; g < 8; ++g) {
    const int k0 = ((g & 4) ? 16 : 0) + 8 * hi + 2 * (g & 3);
    bf16x2 p = *(const bf16x2*)(rowptr + k0);   // 4-byte aligned pair load
    f[2 * g]     = p[0];
    f[2 * g + 1] = p[1];
  }
  return f;
}

// ---------------------------------------------------------------------------
// idx[b][k] = sorted positions where mask is true (exactly MK_ per batch).
// ---------------------------------------------------------------------------
__global__ void build_idx_kernel(const uint8_t* __restrict__ mask,
                                 int* __restrict__ idx) {
  const int b = threadIdx.x;
  if (b >= B_) return;
  int c = 0;
  for (int l = 0; l < L_; ++l) {
    if (mask[(size_t)b * L_ + l]) {
      if (c < MK_) idx[b * MK_ + c] = l;
      ++c;
    }
  }
}

__global__ void f32_to_bf16_kernel(const float* __restrict__ src,
                                   bf16_t* __restrict__ dst, int n) {
  const int i = blockIdx.x * blockDim.x + threadIdx.x;
  if (i < n) dst[i] = (bf16_t)src[i];
}

// ---------------------------------------------------------------------------
// Register key/value row: Kall[b][1024][n] = register[b] . Wr[n] + br[n];
// rows 1025..1055 zeroed so padded WMMA key tiles stay finite.
// ---------------------------------------------------------------------------
__global__ __launch_bounds__(256)
void reg_kv_kernel(const float* __restrict__ reg, const float* __restrict__ Wr,
                   const float* __restrict__ br, bf16_t* __restrict__ Kall) {
  const int b = blockIdx.y;
  const int n = blockIdx.x * 256 + threadIdx.x;
  const float* rr = reg + (size_t)b * D_;
  const float* wr = Wr + (size_t)n * D_;
  float s = br[n];
  for (int d = 0; d < D_; ++d) s += rr[d] * wr[d];
  bf16_t* base = Kall + ((size_t)b * KROWS + MK_) * D_ + n;
  base[0] = (bf16_t)s;
  for (int kr = 1; kr < KROWS - MK_; ++kr) base[(size_t)kr * D_] = (bf16_t)0.0f;
}

// ---------------------------------------------------------------------------
// WMMA GEMM: C[b,m,n] = sum_k A[b,m,k] * W[n,k] + bias[n]
//   AMODE 0: A rows gathered from x (f32) via idx -> bf16 in LDS (vector path)
//   AMODE 1: A is dense bf16 [B][MK_][D_]; staged by double-buffered TDM
//   OMODE 0: store bf16 to outB (row m, per-batch stride outB_bstride)
//   OMODE 1: store f32 scattered to outF[b, idx[b][m], n]
// W tiles double-buffered via TDM: wave 0 posts k+1 DMA before waiting for k
// (per-wave TDM completes in order), hiding DMA latency behind the WMMAs.
// Block: 256 threads = 8 waves; block tile 128(M)x128(N); wave tile 32x64.
// ---------------------------------------------------------------------------
template <int AMODE, int OMODE>
__global__ __launch_bounds__(256)
void gemm_wmma_kernel(const float* __restrict__ xA,
                      const bf16_t* __restrict__ bA,
                      const int* __restrict__ idx,
                      const bf16_t* __restrict__ W,
                      const float* __restrict__ bias,
                      bf16_t* __restrict__ outB, size_t outB_bstride,
                      float* __restrict__ outF) {
  __shared__ __align__(16) bf16_t At[2][128 * 32];
  __shared__ __align__(16) bf16_t Wt[2][128 * 32];

  const int b     = blockIdx.z;
  const int nbase = blockIdx.x * 128;
  const int mbase = blockIdx.y * 128;
  const int tid   = threadIdx.x;
  const int lane  = tid & 31;
  const int wv    = tid >> 5;
  const int wmo   = (wv >> 1) * 32;  // wave M offset: 0/32/64/96
  const int wno   = (wv & 1) * 64;   // wave N offset: 0/64
  const int lo    = lane & 15;
  const int hi    = lane >> 4;

  const int ldrow  = tid >> 1;  // 0..127
  const int ldhalf = tid & 1;   // 16-column half

  v8f acc[2][4];
  const v8f vzero = {0.f, 0.f, 0.f, 0.f, 0.f, 0.f, 0.f, 0.f};
#pragma unroll
  for (int i = 0; i < 2; ++i)
#pragma unroll
    for (int j = 0; j < 4; ++j) acc[i][j] = vzero;

  const int arow_idx = (AMODE == 0) ? idx[b * MK_ + mbase + ldrow] : 0;
  const unsigned wt_lds[2] = {lds_offset_of(&Wt[0][0]), lds_offset_of(&Wt[1][0])};
  const unsigned at_lds[2] = {lds_offset_of(&At[0][0]), lds_offset_of(&At[1][0])};
  const bf16_t* wbase = W + (size_t)nbase * D_;
  const bf16_t* abase = (AMODE == 1) ? bA + ((size_t)b * MK_ + mbase) * D_
                                     : nullptr;
  const int NKT = D_ / 32;

  if (wv == 0) {  // prime the pipeline: tiles for kt = 0
    tdm_load_tile_bf16(wt_lds[0], wbase, 32u, 128u, (unsigned long long)D_);
    if (AMODE == 1)
      tdm_load_tile_bf16(at_lds[0], abase, 32u, 128u, (unsigned long long)D_);
  }

  for (int kt = 0; kt < NKT; ++kt) {
    const int cur = kt & 1;
    if (AMODE == 0) {  // gathered rows: vector loads + f32->bf16 convert
      const float* src =
          xA + ((size_t)b * L_ + arow_idx) * D_ + kt * 32 + ldhalf * 16;
      bf16_t* dst = &At[cur][ldrow * 32 + ldhalf * 16];
#pragma unroll
      for (int i = 0; i < 16; ++i) dst[i] = (bf16_t)src[i];
      if (kt + 1 < NKT) __builtin_prefetch(src + 32, 0, 0);
    }
    if (wv == 0) {
      if (kt + 1 < NKT) {  // post k+1 DMAs, then wait only for the k tiles
        tdm_load_tile_bf16(wt_lds[cur ^ 1], wbase + (kt + 1) * 32,
                           32u, 128u, (unsigned long long)D_);
        if (AMODE == 1) {
          tdm_load_tile_bf16(at_lds[cur ^ 1], abase + (kt + 1) * 32,
                             32u, 128u, (unsigned long long)D_);
          __builtin_amdgcn_s_wait_tensorcnt(2);
        } else {
          __builtin_amdgcn_s_wait_tensorcnt(1);
        }
      } else {
        __builtin_amdgcn_s_wait_tensorcnt(0);
      }
    }
    __syncthreads();

    v16bf af[2], bfr[4];
#pragma unroll
    for (int i = 0; i < 2; ++i)
      af[i] = load_frag_row32(&At[cur][(wmo + i * 16 + lo) * 32], hi);
#pragma unroll
    for (int j = 0; j < 4; ++j)
      bfr[j] = load_frag_row32(&Wt[cur][(wno + j * 16 + lo) * 32], hi);
#pragma unroll
    for (int i = 0; i < 2; ++i)
#pragma unroll
      for (int j = 0; j < 4; ++j)
        acc[i][j] = wmma_bf16(af[i], bfr[j], acc[i][j]);
    __syncthreads();
  }

  // ---- epilogue: bias + store ----
#pragma unroll
  for (int j = 0; j < 4; ++j) {
    const int ng = nbase + wno + j * 16 + lo;
    const float bb = bias[ng];
#pragma unroll
    for (int i = 0; i < 2; ++i) {
#pragma unroll
      for (int r = 0; r < 8; ++r) {
        const int m = mbase + wmo + i * 16 + r + 8 * hi;
        const float v = acc[i][j][r] + bb;
        if (OMODE == 0) {
          outB[(size_t)b * outB_bstride + (size_t)m * D_ + ng] = (bf16_t)v;
        } else {
          const int row = idx[b * MK_ + m];
          outF[((size_t)b * L_ + row) * D_ + ng] = v;
        }
      }
    }
  }
}

// ---------------------------------------------------------------------------
// Flash attention: one wave per (batch, head, 16-query tile).
// scores = Q K^T / sqrt(HD); causal over sorted positions (k <= q) plus the
// register key (index 1024); online softmax; O = P V, all WMMA bf16.
// V tiles double-buffered via TDM. Row-max reductions batched across all 8
// rows per xor step (one DS wait per step); row sums kept as per-lane
// partials and reduced once after the key loop.
// ---------------------------------------------------------------------------
__global__ __launch_bounds__(32)
void flash_attn_kernel(const bf16_t* __restrict__ Q,
                       const bf16_t* __restrict__ Kall,
                       const bf16_t* __restrict__ Vall,
                       bf16_t* __restrict__ O) {
  __shared__ __align__(16) bf16_t vtile[2][32 * HD_];
  __shared__ __align__(4)  bf16_t probs[16 * 32];

  const int b = blockIdx.z, h = blockIdx.y, qt = blockIdx.x;
  const int qbase = qt * 16;
  const int lane = threadIdx.x, lo = lane & 15, hi = lane >> 4;
  const float scale = 0.08838834764831845f;  // 1/sqrt(128)

  // Q fragments (held for the whole key loop)
  v16bf qf[4];
  {
    const bf16_t* qrow = Q + ((size_t)b * MK_ + qbase + lo) * D_ + h * HD_;
#pragma unroll
    for (int c = 0; c < 4; ++c) qf[c] = load_frag_row32(qrow + 32 * c, hi);
  }

  const v8f vzero = {0.f, 0.f, 0.f, 0.f, 0.f, 0.f, 0.f, 0.f};
  v8f accv[8];
#pragma unroll
  for (int t = 0; t < 8; ++t) accv[t] = vzero;
  float mrun[8], lsum[8];  // lsum = per-lane partial row sum
#pragma unroll
  for (int r = 0; r < 8; ++r) { mrun[r] = -1e30f; lsum[r] = 0.0f; }

  const unsigned vt_lds[2] = {lds_offset_of(&vtile[0][0]),
                              lds_offset_of(&vtile[1][0])};
  const bf16_t* vbase = Vall + (size_t)b * KROWS * D_ + h * HD_;

  const int nfull = (qbase + 16 + 31) >> 5;  // 32-key tiles in causal span
  // prefetch tile 0 (keys 0..31)
  tdm_load_tile_bf16(vt_lds[0], vbase, (unsigned)HD_, 32u,
                     (unsigned long long)D_);

  for (int kt = 0; kt <= nfull; ++kt) {
    const int keybase = (kt == nfull) ? MK_ : kt * 32;  // last tile = register
    const int cur = kt & 1;

    __builtin_amdgcn_s_wait_tensorcnt(0);  // V tile `cur` landed
    if (kt < nfull) {                      // post DMA for next tile
      const int nb = (kt + 1 == nfull) ? MK_ : (kt + 1) * 32;
      tdm_load_tile_bf16(vt_lds[cur ^ 1], vbase + (size_t)nb * D_,
                         (unsigned)HD_, 32u, (unsigned long long)D_);
    }
    const bf16_t* vt = &vtile[cur][0];

    // scores: two 16x16 C fragments (keys 0-15, 16-31 of the tile)
    v8f s0 = vzero, s1 = vzero;
    {
      const bf16_t* k0row =
          Kall + ((size_t)b * KROWS + keybase + lo) * D_ + h * HD_;
      const bf16_t* k1row = k0row + (size_t)16 * D_;
#pragma unroll
      for (int c = 0; c < 4; ++c) {
        v16bf bk0 = load_frag_row32(k0row + 32 * c, hi);
        v16bf bk1 = load_frag_row32(k1row + 32 * c, hi);
        s0 = wmma_bf16(qf[c], bk0, s0);
        s1 = wmma_bf16(qf[c], bk1, s1);
      }
    }

    // scale + causal mask, per-row tile max
    const int kidx0 = keybase + lo;
    const int kidx1 = keybase + 16 + lo;
    float sc0[8], sc1[8], mx[8];
#pragma unroll
    for (int r = 0; r < 8; ++r) {
      const int qidx = qbase + r + 8 * hi;
      float v0 = s0[r] * scale, v1 = s1[r] * scale;
      if (!(kidx0 <= qidx || kidx0 == MK_)) v0 = -1e9f;
      if (!(kidx1 <= qidx || kidx1 == MK_)) v1 = -1e9f;
      sc0[r] = v0; sc1[r] = v1;
      mx[r] = fmaxf(v0, v1);
    }
    // batched xor-reduce over the 16 lanes of each half (8 independent rows
    // per step -> one DS wait per step instead of per row)
#pragma unroll
    for (int off = 1; off < 16; off <<= 1) {
#pragma unroll
      for (int r = 0; r < 8; ++r)
        mx[r] = fmaxf(mx[r], __shfl_xor(mx[r], off, 16));
    }
    // online-softmax update; probabilities to LDS in C layout
#pragma unroll
    for (int r = 0; r < 8; ++r) {
      const float mnew = fmaxf(mrun[r], mx[r]);
      const float alpha = __expf(mrun[r] - mnew);
      mrun[r] = mnew;
      const float p0 = __expf(sc0[r] - mnew), p1 = __expf(sc1[r] - mnew);
      lsum[r] = lsum[r] * alpha + p0 + p1;   // per-lane partial (2 keys/lane)
#pragma unroll
      for (int t = 0; t < 8; ++t) accv[t][r] *= alpha;
      probs[(r + 8 * hi) * 32 + lo]      = (bf16_t)p0;
      probs[(r + 8 * hi) * 32 + 16 + lo] = (bf16_t)p1;
    }
    __syncthreads();

    // P (A-layout, 16x32) from LDS; V fragments (B-layout) from vtile[cur]
    v16bf pa = load_frag_row32(&probs[lo * 32], hi);
#pragma unroll
    for (int t = 0; t < 8; ++t) {
      v16bf vb;
#pragma unroll
      for (int g = 0; g < 8; ++g) {
        const int k0 = ((g & 4) ? 16 : 0) + 8 * hi + 2 * (g & 3);
        vb[2 * g]     = vt[(k0)     * HD_ + t * 16 + lo];
        vb[2 * g + 1] = vt[(k0 + 1) * HD_ + t * 16 + lo];
      }
      accv[t] = wmma_bf16(pa, vb, accv[t]);
    }
    __syncthreads();
  }

  // final row-sum reduction (once), batched across rows
#pragma unroll
  for (int off = 1; off < 16; off <<= 1) {
#pragma unroll
    for (int r = 0; r < 8; ++r) lsum[r] += __shfl_xor(lsum[r], off, 16);
  }

  // normalize + store O (bf16)
#pragma unroll
  for (int r = 0; r < 8; ++r) {
    const float inv = 1.0f / lsum[r];
    const int q = qbase + r + 8 * hi;
    bf16_t* orow = O + ((size_t)b * MK_ + q) * D_ + h * HD_;
#pragma unroll
    for (int t = 0; t < 8; ++t) orow[t * 16 + lo] = (bf16_t)(accv[t][r] * inv);
  }
}

// ---------------------------------------------------------------------------
// Launcher
// ---------------------------------------------------------------------------
extern "C" void kernel_launch(void* const* d_in, const int* in_sizes, int n_in,
                              void* d_out, int out_size, void* d_ws, size_t ws_size,
                              hipStream_t stream) {
  const float*   x    = (const float*)d_in[0];
  const uint8_t* mask = (const uint8_t*)d_in[1];
  const float*   reg  = (const float*)d_in[2];
  const float* Wq = (const float*)d_in[3];  const float* bq  = (const float*)d_in[4];
  const float* Wk = (const float*)d_in[5];  const float* bk  = (const float*)d_in[6];
  const float* Wv = (const float*)d_in[7];  const float* bv  = (const float*)d_in[8];
  const float* Wrk = (const float*)d_in[9];  const float* brk = (const float*)d_in[10];
  const float* Wrv = (const float*)d_in[11]; const float* brv = (const float*)d_in[12];
  const float* Wo = (const float*)d_in[13]; const float* bo  = (const float*)d_in[14];
  float* out = (float*)d_out;

  char* ws = (char*)d_ws;
  size_t off = 0;
  auto alloc = [&](size_t bytes) -> char* {
    char* p = ws + off;
    off += (bytes + 255) & ~(size_t)255;
    return p;
  };
  int*    idx = (int*)alloc((size_t)B_ * MK_ * sizeof(int));
  bf16_t* Wqb = (bf16_t*)alloc((size_t)D_ * D_ * 2);
  bf16_t* Wkb = (bf16_t*)alloc((size_t)D_ * D_ * 2);
  bf16_t* Wvb = (bf16_t*)alloc((size_t)D_ * D_ * 2);
  bf16_t* Wob = (bf16_t*)alloc((size_t)D_ * D_ * 2);
  bf16_t* Qb  = (bf16_t*)alloc((size_t)B_ * MK_ * D_ * 2);
  bf16_t* Kb  = (bf16_t*)alloc((size_t)B_ * KROWS * D_ * 2);
  bf16_t* Vb  = (bf16_t*)alloc((size_t)B_ * KROWS * D_ * 2);
  bf16_t* Ob  = (bf16_t*)alloc((size_t)B_ * MK_ * D_ * 2);

  // out = x (selected rows overwritten by the final scatter GEMM)
  (void)hipMemcpyAsync(out, x, (size_t)B_ * L_ * D_ * sizeof(float),
                       hipMemcpyDeviceToDevice, stream);

  build_idx_kernel<<<1, B_, 0, stream>>>(mask, idx);

  const int nW = D_ * D_;
  f32_to_bf16_kernel<<<(nW + 255) / 256, 256, 0, stream>>>(Wq, Wqb, nW);
  f32_to_bf16_kernel<<<(nW + 255) / 256, 256, 0, stream>>>(Wk, Wkb, nW);
  f32_to_bf16_kernel<<<(nW + 255) / 256, 256, 0, stream>>>(Wv, Wvb, nW);
  f32_to_bf16_kernel<<<(nW + 255) / 256, 256, 0, stream>>>(Wo, Wob, nW);

  const dim3 gg(D_ / 128, MK_ / 128, B_);  // (16, 8, 8)
  gemm_wmma_kernel<0, 0><<<gg, 256, 0, stream>>>(x, nullptr, idx, Wqb, bq,
                                                 Qb, (size_t)MK_ * D_, nullptr);
  gemm_wmma_kernel<0, 0><<<gg, 256, 0, stream>>>(x, nullptr, idx, Wkb, bk,
                                                 Kb, (size_t)KROWS * D_, nullptr);
  gemm_wmma_kernel<0, 0><<<gg, 256, 0, stream>>>(x, nullptr, idx, Wvb, bv,
                                                 Vb, (size_t)KROWS * D_, nullptr);

  reg_kv_kernel<<<dim3(D_ / 256, B_), 256, 0, stream>>>(reg, Wrk, brk, Kb);
  reg_kv_kernel<<<dim3(D_ / 256, B_), 256, 0, stream>>>(reg, Wrv, brv, Vb);

  flash_attn_kernel<<<dim3(MK_ / 16, H_, B_), 32, 0, stream>>>(Qb, Kb, Vb, Ob);

  gemm_wmma_kernel<1, 1><<<gg, 256, 0, stream>>>(nullptr, Ob, idx, Wob, bo,
                                                 nullptr, 0, out);
}
